// DialogueEIN_73057393705571
// MI455X (gfx1250) — matrix operand order repaired and checked by reading, not compiled, not run on hardware
//
#include <hip/hip_runtime.h>

#define B_  64
#define T_  256
#define H_  512
#define NH_ 8
#define DH_ 64
#define E_  7
#define NT_ (B_ * T_)
#define NEGB (-10000.0f)

typedef __attribute__((ext_vector_type(16))) _Float16 v16h;
typedef __attribute__((ext_vector_type(8)))  _Float16 v8h;
typedef __attribute__((ext_vector_type(8)))  float    v8f;

// ---------------------------------------------------------------------------
// WMMA fragment helpers (CDNA5 16x16x32 f16, wave32; contiguous k order, same for A and B)
// A (16x32, row-major source, 16B-aligned): lane l holds row M=l&15;
//   elements e -> K = (l>>4)*16 + e  ->  lo = p[0..7], hi = p[8..15]
// B (32x16, from N-major/transposed source): lane l holds col N=l&15;
//   elements e -> K = (l>>4)*16 + e  ->  16 contiguous halves
// D (f32 16x16): lane l, vgpr v -> M = (l>>4)*8 + v, N = l&15
// ---------------------------------------------------------------------------
__device__ __forceinline__ v16h ld_frag_a(const _Float16* p) {
  v8h lo = *(const v8h*)p;
  v8h hi = *(const v8h*)(p + 8);
  v16h r;
#pragma unroll
  for (int e = 0; e < 8; ++e) { r[e] = lo[e]; r[e + 8] = hi[e]; }
  return r;
}

__device__ __forceinline__ v16h ld_frag_b(const _Float16* p) {
  v8h lo = *(const v8h*)p;
  v8h hi = *(const v8h*)(p + 8);
  v16h r;
#pragma unroll
  for (int e = 0; e < 8; ++e) { r[e] = lo[e]; r[e + 8] = hi[e]; }
  return r;
}

__device__ __forceinline__ v8f wmma_f16(v16h a, v16h b, v8f c) {
  return __builtin_amdgcn_wmma_f32_16x16x32_f16(false, a, false, b, (short)0, c,
                                                false, false);
}

// ---------------------------------------------------------------------------
// f32 -> f16 elementwise convert
// ---------------------------------------------------------------------------
__global__ void cvt16_k(const float* __restrict__ X, _Float16* __restrict__ Y, int n) {
  int i = blockIdx.x * blockDim.x + threadIdx.x;
  if (i < n) Y[i] = (_Float16)X[i];
}

// ---------------------------------------------------------------------------
// Weight prep: W [K,N] f32 row-major -> Wt [N,K] f16 (so B-fragments are
// contiguous 32B loads)
// ---------------------------------------------------------------------------
__global__ void prep_w_k(const float* __restrict__ W, _Float16* __restrict__ Wt,
                         int Kd, int Nd) {
  int i = blockIdx.x * blockDim.x + threadIdx.x;
  if (i >= Kd * Nd) return;
  int k = i / Nd, n = i % Nd;
  Wt[(size_t)n * Kd + k] = (_Float16)W[i];
}

// ---------------------------------------------------------------------------
// WMMA GEMM: C[M,N] = A[M,K](f16) @ Wt[N,K](f16)^T + bias + res
// block = 128 threads (4 waves), 64x64 tile; wave -> 16x64 stripe (4 tiles)
// ---------------------------------------------------------------------------
__global__ __launch_bounds__(128) void gemm16_k(
    const _Float16* __restrict__ A, const _Float16* __restrict__ Wt,
    const float* __restrict__ bias, const float* __restrict__ res,
    float* __restrict__ Cf, _Float16* __restrict__ Ch,
    int M, int N, int K, int ldch, int col0)
{
  const int m0   = blockIdx.x * 64;
  const int n0   = blockIdx.y * 64;
  const int wave = threadIdx.x >> 5;
  const int lane = threadIdx.x & 31;
  const int nl   = lane & 15;
  const int g    = lane >> 4;
  const int mrow = m0 + wave * 16;

  const _Float16* ap = A  + (size_t)(mrow + nl) * K + g * 16;
  const _Float16* bp = Wt + (size_t)(n0 + nl) * K + g * 16;

  const v8f zero = {0.f, 0.f, 0.f, 0.f, 0.f, 0.f, 0.f, 0.f};
  v8f acc[4];
#pragma unroll
  for (int t = 0; t < 4; ++t) acc[t] = zero;

  for (int k0 = 0; k0 < K; k0 += 32) {
    v16h a = ld_frag_a(ap + k0);
#pragma unroll
    for (int t = 0; t < 4; ++t) {
      v16h b = ld_frag_b(bp + (size_t)t * 16 * K + k0);
      acc[t] = wmma_f16(a, b, acc[t]);
    }
  }

  const int mm = mrow + g * 8;
#pragma unroll
  for (int t = 0; t < 4; ++t) {
    const int n = n0 + t * 16 + nl;
    const float bv = bias ? bias[n] : 0.0f;
#pragma unroll
    for (int v = 0; v < 8; ++v) {
      const size_t m = (size_t)(mm + v);
      float val = acc[t][v] + bv;
      if (res) val += res[m * N + n];
      if (Cf) Cf[m * N + n] = val;
      if (Ch) Ch[m * ldch + col0 + n] = (_Float16)val;
    }
  }
}

// ---------------------------------------------------------------------------
// emo K/V projections: k_emo/v_emo [E,H] = emo_table @ t_Wk/t_Wv + bias (f32)
// ---------------------------------------------------------------------------
__global__ void emo_proj_k(const float* __restrict__ emo,
                           const float* __restrict__ Wk, const float* __restrict__ bk,
                           const float* __restrict__ Wv, const float* __restrict__ bv,
                           float* __restrict__ kE, float* __restrict__ vE)
{
  const int idx = blockIdx.x * blockDim.x + threadIdx.x;
  if (idx >= 2 * E_ * H_) return;
  const int which = idx / (E_ * H_);
  const int r = idx % (E_ * H_);
  const int e = r / H_, n = r % H_;
  const float* W = which ? Wv : Wk;
  float acc = (which ? bv : bk)[n];
  for (int k = 0; k < H_; ++k) acc += emo[e * H_ + k] * W[(size_t)k * H_ + n];
  (which ? vE : kE)[r] = acc;
}

// ---------------------------------------------------------------------------
// Stage-1 attention over E=7 emotion tokens (tiny: scalar per (b,t,head))
// ---------------------------------------------------------------------------
__global__ __launch_bounds__(256) void attn_emo_k(
    const _Float16* __restrict__ Q, const float* __restrict__ kE,
    const float* __restrict__ vE, _Float16* __restrict__ Ctx)
{
  const int idx = blockIdx.x * blockDim.x + threadIdx.x;
  if (idx >= NT_ * NH_) return;
  const int h  = idx & (NH_ - 1);
  const int bt = idx >> 3;
  const _Float16* q = Q + (size_t)bt * H_ + h * DH_;
  float qf[DH_];
#pragma unroll
  for (int d = 0; d < DH_; ++d) qf[d] = (float)q[d];
  float s[E_]; float mx = -3.0e38f;
#pragma unroll
  for (int e = 0; e < E_; ++e) {
    const float* kr = kE + (size_t)e * H_ + h * DH_;
    float acc = 0.f;
    for (int d = 0; d < DH_; ++d) acc += qf[d] * kr[d];
    s[e] = acc * 0.125f;   // 1/sqrt(64)
    mx = fmaxf(mx, s[e]);
  }
  float sum = 0.f;
#pragma unroll
  for (int e = 0; e < E_; ++e) { s[e] = __expf(s[e] - mx); sum += s[e]; }
  const float inv = 1.f / sum;
  _Float16* out = Ctx + (size_t)bt * H_ + h * DH_;
  for (int d = 0; d < DH_; ++d) {
    float acc = 0.f;
#pragma unroll
    for (int e = 0; e < E_; ++e) acc += s[e] * vE[(size_t)e * H_ + h * DH_ + d];
    out[d] = (_Float16)(acc * inv);
  }
}

// ---------------------------------------------------------------------------
// LayerNorm over H=512, eps=1e-12; optional f32 + f16 outputs
// ---------------------------------------------------------------------------
__global__ __launch_bounds__(256) void layernorm_k(
    const float* __restrict__ X, const float* __restrict__ gam,
    const float* __restrict__ bet, float* __restrict__ Yf, _Float16* __restrict__ Yh)
{
  __shared__ float red[256];
  const int row = blockIdx.x;
  const int tid = threadIdx.x;
  const float* xr = X + (size_t)row * H_;
  const float a = xr[tid];
  const float b = xr[tid + 256];
  red[tid] = a + b;
  __syncthreads();
  for (int s = 128; s > 0; s >>= 1) { if (tid < s) red[tid] += red[tid + s]; __syncthreads(); }
  const float mean = red[0] * (1.0f / H_);
  __syncthreads();
  const float d0 = a - mean, d1 = b - mean;
  red[tid] = d0 * d0 + d1 * d1;
  __syncthreads();
  for (int s = 128; s > 0; s >>= 1) { if (tid < s) red[tid] += red[tid + s]; __syncthreads(); }
  const float inv = rsqrtf(red[0] * (1.0f / H_) + 1e-12f);
  const float y0 = d0 * inv * gam[tid] + bet[tid];
  const float y1 = d1 * inv * gam[tid + 256] + bet[tid + 256];
  if (Yf) { Yf[(size_t)row * H_ + tid] = y0; Yf[(size_t)row * H_ + tid + 256] = y1; }
  if (Yh) { Yh[(size_t)row * H_ + tid] = (_Float16)y0; Yh[(size_t)row * H_ + tid + 256] = (_Float16)y1; }
}

// ---------------------------------------------------------------------------
// V transpose: V [b*T+t, h*64+d] -> Vt [(b*NH+h)*64+d, t]  (for B-fragments)
// ---------------------------------------------------------------------------
__global__ void transpose_v_k(const _Float16* __restrict__ V, _Float16* __restrict__ Vt) {
  int idx = blockIdx.x * blockDim.x + threadIdx.x;
  if (idx >= NT_ * H_) return;
  int col = idx & (H_ - 1);
  int row = idx >> 9;
  int b = row >> 8, t = row & 255;
  int h = col >> 6, d = col & 63;
  Vt[(((size_t)(b * NH_ + h)) * DH_ + d) * T_ + t] = V[idx];
}

// ---------------------------------------------------------------------------
// Masked branch attention. grid=(B*NH, T/64), block=128 (4 waves).
// Each wave owns a 16x256 score tile in 16 WMMA accumulators; mask bias is
// computed on the fly; softmax via __shfl_xor within 16-lane groups; P staged
// in per-wave 8KB LDS then PV via WMMA against transposed V. No barriers.
// ---------------------------------------------------------------------------
__global__ __launch_bounds__(128) void attn_branch_k(
    const _Float16* __restrict__ Q, const _Float16* __restrict__ K,
    const _Float16* __restrict__ Vt, const int* __restrict__ lengths,
    const int* __restrict__ speakers, _Float16* __restrict__ Ctx, int branch)
{
  __shared__ _Float16 sP[4 * 16 * 256];  // 32 KB total, 8 KB per wave
  const int b    = blockIdx.x / NH_;
  const int h    = blockIdx.x % NH_;
  const int q0   = blockIdx.y * 64;
  const int wave = threadIdx.x >> 5;
  const int lane = threadIdx.x & 31;
  const int nl   = lane & 15;
  const int g    = lane >> 4;

  // ---- S = Q @ K^T * 1/8 ------------------------------------------------
  const _Float16* ap = Q + (size_t)(b * T_ + q0 + wave * 16 + nl) * H_ + h * DH_ + g * 16;
  v16h a0 = ld_frag_a(ap);
  v16h a1 = ld_frag_a(ap + 32);

  const _Float16* kp = K + (size_t)(b * T_ + nl) * H_ + h * DH_ + g * 16;
  const v8f zero = {0.f, 0.f, 0.f, 0.f, 0.f, 0.f, 0.f, 0.f};
  v8f acc[16];
#pragma unroll
  for (int nt = 0; nt < 16; ++nt) {
    v8f c = zero;
    const _Float16* bp = kp + (size_t)nt * 16 * H_;
    c = wmma_f16(a0, ld_frag_b(bp), c);
    c = wmma_f16(a1, ld_frag_b(bp + 32), c);
    acc[nt] = c;
  }

  // ---- mask bias + softmax (row m = g*8+v lives in this half-wave) -------
  const int Lb = lengths[b];
  const int jb = q0 + wave * 16 + g * 8;
  bool rv[8]; int spj[8];
#pragma unroll
  for (int v = 0; v < 8; ++v) {
    const int j = jb + v;
    rv[v] = j < Lb;
    spj[v] = speakers[b * T_ + j];
  }
  bool cv[16]; int spk[16];
#pragma unroll
  for (int nt = 0; nt < 16; ++nt) {
    const int kk = nt * 16 + nl;
    cv[nt] = kk < Lb;
    spk[nt] = speakers[b * T_ + kk];
  }
  float mx[8];
#pragma unroll
  for (int v = 0; v < 8; ++v) mx[v] = -3.0e38f;
#pragma unroll
  for (int nt = 0; nt < 16; ++nt) {
    const int kk = nt * 16 + nl;
#pragma unroll
    for (int v = 0; v < 8; ++v) {
      int dj = (jb + v) - kk; if (dj < 0) dj = -dj;
      float bias;
      if (branch == 0)      bias = cv[nt] ? 0.f : NEGB;
      else if (branch == 1) bias = (rv[v] && cv[nt] && dj <= 2) ? 0.f : NEGB;
      else if (branch == 2) bias = (rv[v] && (spk[nt] == spj[v])) ? 0.f : NEGB;
      else                  bias = (rv[v] && (spk[nt] != spj[v]) && cv[nt]) ? 0.f : NEGB;
      const float s = acc[nt][v] * 0.125f + bias;
      acc[nt][v] = s;
      mx[v] = fmaxf(mx[v], s);
    }
  }
#pragma unroll
  for (int v = 0; v < 8; ++v) {
#pragma unroll
    for (int off = 1; off < 16; off <<= 1)
      mx[v] = fmaxf(mx[v], __shfl_xor(mx[v], off, 32));
  }
  float sum[8];
#pragma unroll
  for (int v = 0; v < 8; ++v) sum[v] = 0.f;
#pragma unroll
  for (int nt = 0; nt < 16; ++nt) {
#pragma unroll
    for (int v = 0; v < 8; ++v) {
      const float e = __expf(acc[nt][v] - mx[v]);
      acc[nt][v] = e;
      sum[v] += e;
    }
  }
#pragma unroll
  for (int v = 0; v < 8; ++v) {
#pragma unroll
    for (int off = 1; off < 16; off <<= 1)
      sum[v] += __shfl_xor(sum[v], off, 32);
    sum[v] = 1.f / sum[v];
  }
  _Float16* sPw = sP + wave * 16 * 256;
#pragma unroll
  for (int nt = 0; nt < 16; ++nt) {
#pragma unroll
    for (int v = 0; v < 8; ++v)
      sPw[(g * 8 + v) * 256 + nt * 16 + nl] = (_Float16)(acc[nt][v] * sum[v]);
  }

  // ---- ctx = P(16x256) @ V(256x64) --------------------------------------
  const _Float16* pap = sPw + (size_t)nl * 256 + g * 16;
#pragma unroll
  for (int nt2 = 0; nt2 < 4; ++nt2) {
    v8f c = zero;
    const _Float16* vb =
        Vt + ((size_t)((b * NH_ + h) * DH_ + nt2 * 16 + nl)) * T_ + g * 16;
#pragma unroll
    for (int k0 = 0; k0 < 256; k0 += 32) {
      v16h a  = ld_frag_a(pap + k0);
      v16h bb = ld_frag_b(vb + k0);
      c = wmma_f16(a, bb, c);
    }
    const size_t orow = (size_t)(b * T_ + q0 + wave * 16 + g * 8);
    const int ocol = h * DH_ + nt2 * 16 + nl;
#pragma unroll
    for (int v = 0; v < 8; ++v)
      Ctx[(orow + v) * H_ + ocol] = (_Float16)c[v];
  }
}

// ---------------------------------------------------------------------------
// Host-side orchestration
// ---------------------------------------------------------------------------
extern "C" void kernel_launch(void* const* d_in, const int* in_sizes, int n_in,
                              void* d_out, int out_size, void* d_ws, size_t ws_size,
                              hipStream_t stream) {
  (void)in_sizes; (void)n_in; (void)out_size; (void)ws_size;

  const float* x        = (const float*)d_in[0];
  const int*   lengths  = (const int*)d_in[1];
  const int*   speakers = (const int*)d_in[2];
  const float* emo      = (const float*)d_in[3];
  const float* t_Wq = (const float*)d_in[4];  const float* t_bq = (const float*)d_in[5];
  const float* t_Wk = (const float*)d_in[6];  const float* t_bk = (const float*)d_in[7];
  const float* t_Wv = (const float*)d_in[8];  const float* t_bv = (const float*)d_in[9];
  const float* t_Wo = (const float*)d_in[10]; const float* t_bo = (const float*)d_in[11];
  const float* t_ln_g = (const float*)d_in[12]; const float* t_ln_b = (const float*)d_in[13];
  const float* b_Wq = (const float*)d_in[14]; const float* b_bq = (const float*)d_in[15];
  const float* b_Wk = (const float*)d_in[16]; const float* b_bk = (const float*)d_in[17];
  const float* b_Wv = (const float*)d_in[18]; const float* b_bv = (const float*)d_in[19];
  const float* b_Wo = (const float*)d_in[20]; const float* b_bo = (const float*)d_in[21];
  const float* W1 = (const float*)d_in[22];   const float* b1 = (const float*)d_in[23];
  const float* ln2_g = (const float*)d_in[24]; const float* ln2_b = (const float*)d_in[25];
  const float* W2 = (const float*)d_in[26];

  char* ws = (char*)d_ws;
  size_t off = 0;
  auto take = [&](size_t bytes) -> char* {
    char* p = ws + off;
    off += (bytes + 255) & ~(size_t)255;
    return p;
  };
  _Float16* x16   = (_Float16*)take((size_t)NT_ * H_ * 2);
  _Float16* q16   = (_Float16*)take((size_t)NT_ * H_ * 2);
  _Float16* k16   = (_Float16*)take((size_t)NT_ * H_ * 2);
  _Float16* v16   = (_Float16*)take((size_t)NT_ * H_ * 2);
  _Float16* vt16  = (_Float16*)take((size_t)NT_ * H_ * 2);
  _Float16* a16   = (_Float16*)take((size_t)NT_ * H_ * 2);
  _Float16* ht16  = (_Float16*)take((size_t)NT_ * H_ * 2);
  _Float16* he16  = (_Float16*)take((size_t)NT_ * H_ * 2);
  _Float16* cat16 = (_Float16*)take((size_t)NT_ * 4 * H_ * 2);
  float*    pre32 = (float*)take((size_t)NT_ * H_ * 4);
  float*    ht32  = (float*)take((size_t)NT_ * H_ * 4);
  _Float16* wt    = (_Float16*)take((size_t)4 * H_ * H_ * 2);  // weight scratch (up to 2048x512)
  float*    kemo  = (float*)take((size_t)E_ * H_ * 4);
  float*    vemo  = (float*)take((size_t)E_ * H_ * 4);

  const dim3 gGemm(NT_ / 64, H_ / 64);        // (256, 8)
  const dim3 gAttn(B_ * NH_, T_ / 64);        // (512, 4)
  const int nElem = NT_ * H_;                 // 8.4M
  const int eBlocks = (nElem + 255) / 256;
  const int wBlocks512 = (H_ * H_ + 255) / 256;

  // 1) x -> f16
  cvt16_k<<<eBlocks, 256, 0, stream>>>(x, x16, nElem);

  // 2) q = x @ t_Wq + t_bq   (f16 out)
  prep_w_k<<<wBlocks512, 256, 0, stream>>>(t_Wq, wt, H_, H_);
  gemm16_k<<<gGemm, 128, 0, stream>>>(x16, wt, t_bq, nullptr, nullptr, q16,
                                      NT_, H_, H_, H_, 0);

  // 3) emo K/V projections (f32, tiny)
  emo_proj_k<<<(2 * E_ * H_ + 255) / 256, 256, 0, stream>>>(
      emo, t_Wk, t_bk, t_Wv, t_bv, kemo, vemo);

  // 4) stage-1 attention over 7 emotion tokens -> a16
  attn_emo_k<<<(NT_ * NH_) / 256, 256, 0, stream>>>(q16, kemo, vemo, a16);

  // 5) pre = ctx @ t_Wo + t_bo + x  -> LN -> h_t (f32 + f16)
  prep_w_k<<<wBlocks512, 256, 0, stream>>>(t_Wo, wt, H_, H_);
  gemm16_k<<<gGemm, 128, 0, stream>>>(a16, wt, t_bo, x, pre32, nullptr,
                                      NT_, H_, H_, H_, 0);
  layernorm_k<<<NT_, 256, 0, stream>>>(pre32, t_ln_g, t_ln_b, ht32, ht16);

  // 6) four masked self-attention branches -> cat16
  for (int i = 0; i < 4; ++i) {
    const size_t wo = (size_t)i * H_ * H_;
    prep_w_k<<<wBlocks512, 256, 0, stream>>>(b_Wq + wo, wt, H_, H_);
    gemm16_k<<<gGemm, 128, 0, stream>>>(ht16, wt, b_bq + i * H_, nullptr,
                                        nullptr, q16, NT_, H_, H_, H_, 0);
    prep_w_k<<<wBlocks512, 256, 0, stream>>>(b_Wk + wo, wt, H_, H_);
    gemm16_k<<<gGemm, 128, 0, stream>>>(ht16, wt, b_bk + i * H_, nullptr,
                                        nullptr, k16, NT_, H_, H_, H_, 0);
    prep_w_k<<<wBlocks512, 256, 0, stream>>>(b_Wv + wo, wt, H_, H_);
    gemm16_k<<<gGemm, 128, 0, stream>>>(ht16, wt, b_bv + i * H_, nullptr,
                                        nullptr, v16, NT_, H_, H_, H_, 0);
    transpose_v_k<<<eBlocks, 256, 0, stream>>>(v16, vt16);
    attn_branch_k<<<gAttn, 128, 0, stream>>>(q16, k16, vt16, lengths, speakers,
                                             a16, i);
    prep_w_k<<<wBlocks512, 256, 0, stream>>>(b_Wo + wo, wt, H_, H_);
    gemm16_k<<<gGemm, 128, 0, stream>>>(a16, wt, b_bo + i * H_, nullptr,
                                        nullptr, cat16, NT_, H_, H_,
                                        4 * H_, i * H_);
  }

  // 7) pre = cat @ W1 + b1 + h_t  -> LN -> h_e (f16)
  prep_w_k<<<(4 * H_ * H_ + 255) / 256, 256, 0, stream>>>(W1, wt, 4 * H_, H_);
  gemm16_k<<<gGemm, 128, 0, stream>>>(cat16, wt, b1, ht32, pre32, nullptr,
                                      NT_, H_, 4 * H_, H_, 0);
  layernorm_k<<<NT_, 256, 0, stream>>>(pre32, ln2_g, ln2_b, nullptr, he16);

  // 8) out = h_e @ W2  (f32 output)
  prep_w_k<<<wBlocks512, 256, 0, stream>>>(W2, wt, H_, H_);
  gemm16_k<<<gGemm, 128, 0, stream>>>(he16, wt, nullptr, nullptr,
                                      (float*)d_out, nullptr, NT_, H_, H_, H_, 0);
}
